// LuongConcatAttention_45698452030203
// MI455X (gfx1250) — matrix operand order repaired and checked by reading, not compiled, not run on hardware
//
#include <hip/hip_runtime.h>
#include <hip/hip_bf16.h>
#include <math.h>

#define B_   32
#define S_   2048
#define D_   1024   // DD == DE == DA == 1024
#define NEGV -1000000000.0f
#define LDSK 1032   // padded LDS row stride in halves (bank-conflict free)

typedef __attribute__((ext_vector_type(16))) __bf16 v16bf;
typedef __attribute__((ext_vector_type(2)))  __bf16 v2bf;
typedef __attribute__((ext_vector_type(8)))  float  v8f;

union FragU { v16bf v; uint4 q[2]; };

__device__ __forceinline__ unsigned short f2bf(float f) {
    union { float f; unsigned u; } x; x.f = f;
    unsigned r = x.u + 0x7FFFu + ((x.u >> 16) & 1u);   // round-to-nearest-even
    return (unsigned short)(r >> 16);
}

// pack two f32 -> one dword of bf16x2 (hardware v_cvt_pk_bf16_f32 when available)
__device__ __forceinline__ unsigned pack2bf(float lo, float hi) {
#if defined(__AMDGCN__) && __has_builtin(__builtin_amdgcn_cvt_pk_bf16_f32)
    v2bf t = __builtin_amdgcn_cvt_pk_bf16_f32(lo, hi);
    unsigned u; __builtin_memcpy(&u, &t, 4); return u;
#else
    return (unsigned)f2bf(lo) | ((unsigned)f2bf(hi) << 16);
#endif
}

// hardware v_tanh_f32 (single TRANS op) when available
__device__ __forceinline__ float fast_tanh(float x) {
#if defined(__AMDGCN__) && __has_builtin(__builtin_amdgcn_tanhf)
    return __builtin_amdgcn_tanhf(x);
#else
    return tanhf(x);
#endif
}

// ---------------------------------------------------------------- kernel 0:
// We (f32, columns DD..DD+DE of W_c) -> bf16 into ws, row-major [a][k]
__global__ __launch_bounds__(256) void conv_we(const float* __restrict__ Wc,
                                               unsigned* __restrict__ WeB) {
    int i = blockIdx.x * 256 + threadIdx.x;     // 512K dwords (1M bf16)
    int a = i >> 9, k = (i & 511) * 2;
    const float* src = Wc + (size_t)a * 2048 + 1024 + k;
    WeB[i] = pack2bf(src[0], src[1]);
}

// ---------------------------------------------------------------- kernel 1:
// q_proj[b][a] = sum_d q[b][d] * W_c[a][d]   (one warp per 'a', grid 128)
__global__ __launch_bounds__(256) void qprojk(const float* __restrict__ q,
                                              const float* __restrict__ Wc,
                                              float* __restrict__ qp) {
    int warp = threadIdx.x >> 5, lane = threadIdx.x & 31;
    int a = blockIdx.x * 8 + warp;
    const float* w = Wc + (size_t)a * 2048;
    for (int b = 0; b < B_; ++b) {
        const float* qb = q + b * D_;
        float acc = 0.0f;
        for (int d = lane; d < D_; d += 32) acc += w[d] * qb[d];
        #pragma unroll
        for (int off = 16; off; off >>= 1) acc += __shfl_xor(acc, off);
        if (lane == 0) qp[b * D_ + a] = acc;
    }
}

// ---------------------------------------------------------------- kernel 2:
// Fused e_proj GEMM (bf16 WMMA, f32 accum) + tanh + V-dot -> raw scores.
// Block: 256 threads = 8 waves; owns 64 flattened (b,s) rows.
__global__ __launch_bounds__(256) void attnk(const float* __restrict__ E,
                                             const unsigned short* __restrict__ WeB,
                                             const float* __restrict__ qp,
                                             const float* __restrict__ Vv,
                                             float* __restrict__ scores) {
    extern __shared__ char smem[];
    unsigned short* aTile = (unsigned short*)smem;              // 64 x LDSK bf16
    float* sred = (float*)(smem + 64 * LDSK * 2);               // 128 floats

    const int tid  = threadIdx.x;
    const int lane = tid & 31, wave = tid >> 5;
    const int M0   = blockIdx.x * 64;          // flattened b*S + s base
    const int b    = M0 >> 11;                 // 64 | 2048 -> one batch per block

    // ---- Phase 1: stage E tile to LDS as bf16 (vectorized, coalesced) ----
    {
        const float* Eb = E + (size_t)M0 * D_;
        #pragma unroll 4
        for (int i = 0; i < 32; ++i) {
            int idx = (i * 256 + tid) * 8;
            int row = idx >> 10, col = idx & 1023;
            const float4* src = (const float4*)(Eb + (size_t)row * D_ + col);
            float4 f0 = src[0], f1 = src[1];
            uint4 pk;
            pk.x = pack2bf(f0.x, f0.y);
            pk.y = pack2bf(f0.z, f0.w);
            pk.z = pack2bf(f1.x, f1.y);
            pk.w = pack2bf(f1.z, f1.w);
            *(uint4*)(aTile + row * LDSK + col) = pk;
        }
    }
    __syncthreads();

    // wave tiling: 4 M-subtiles x 2 N-halves
    const int mr    = (wave & 3) * 16;     // wave's 16 rows within tile
    const int nh    = wave >> 2;           // 0/1: which 32-wide N half of the chunk
    const int lan16 = lane & 15;
    const int hi    = lane >> 4;
    const int kgrpA = hi * 8;              // A frag: lanes16-31 hold K+8 / K+24
    const int kgrpB = hi * 8;              // B frag: lanes16-31 hold K+8 / K+24
    const unsigned short* aBase = aTile + (mr + lan16) * LDSK + kgrpA;

    float ps[8];
    #pragma unroll
    for (int r = 0; r < 8; ++r) ps[r] = 0.0f;
    const float* qpb = qp + b * D_;

    for (int nc = 0; nc < 16; ++nc) {
        const int nbase = nc * 64 + nh * 32;
        const unsigned short* b0p = WeB + (size_t)(nbase + lan16) * D_ + kgrpB;
        const unsigned short* b1p = b0p + 16 * D_;
        v8f acc0 = {}; v8f acc1 = {};
        #pragma unroll 4
        for (int k0 = 0; k0 < D_; k0 += 32) {
            FragU a, b0, b1;
            a.q[0]  = *(const uint4*)(aBase + k0);        // K {0..7} (+8 hi lanes)
            a.q[1]  = *(const uint4*)(aBase + k0 + 16);   // K {16..23} (+8 hi lanes)
            b0.q[0] = *(const uint4*)(b0p + k0);          // K kb..kb+7
            b0.q[1] = *(const uint4*)(b0p + k0 + 16);     // K kb+16..kb+23
            b1.q[0] = *(const uint4*)(b1p + k0);
            b1.q[1] = *(const uint4*)(b1p + k0 + 16);
            acc0 = __builtin_amdgcn_wmma_f32_16x16x32_bf16(
                       false, a.v, false, b0.v, (short)0, acc0, false, false);
            acc1 = __builtin_amdgcn_wmma_f32_16x16x32_bf16(
                       false, a.v, false, b1.v, (short)0, acc1, false, false);
        }
        // epilogue for this N chunk: tanh(C + q_proj) . V
        const int n0 = nbase + lan16;
        float q0 = qpb[n0], q1 = qpb[n0 + 16];
        float v0 = Vv[n0],  v1 = Vv[n0 + 16];
        #pragma unroll
        for (int r = 0; r < 8; ++r)
            ps[r] += v0 * fast_tanh(acc0[r] + q0) + v1 * fast_tanh(acc1[r] + q1);
    }

    // cross-lane reduction: lane groups of 16 share one C-row (lanes>=16 -> M+8)
    #pragma unroll
    for (int r = 0; r < 8; ++r) {
        float v = ps[r];
        v += __shfl_xor(v, 1); v += __shfl_xor(v, 2);
        v += __shfl_xor(v, 4); v += __shfl_xor(v, 8);
        if (lan16 == 0) sred[nh * 64 + mr + r + 8 * hi] = v;  // unique slot per wave
    }
    __syncthreads();
    if (tid < 64) scores[M0 + tid] = sred[tid] + sred[64 + tid];
}

// ---------------------------------------------------------------- kernel 3:
// masked softmax over S, in place in d_out's weights region
__global__ __launch_bounds__(256) void softmaxk(const int* __restrict__ mask,
                                                float* __restrict__ wts) {
    const int b = blockIdx.x, t = threadIdx.x;
    float* sc = wts + (size_t)b * S_;
    const int* mk = mask + (size_t)b * S_;
    __shared__ float red[256];
    float vals[8]; float mx = -INFINITY;
    #pragma unroll
    for (int i = 0; i < 8; ++i) {
        int s = t + i * 256;
        float v = (mk[s] == 0) ? NEGV : sc[s];
        vals[i] = v; mx = fmaxf(mx, v);
    }
    red[t] = mx; __syncthreads();
    for (int off = 128; off; off >>= 1) {
        if (t < off) red[t] = fmaxf(red[t], red[t + off]);
        __syncthreads();
    }
    mx = red[0]; __syncthreads();
    float sum = 0.0f;
    #pragma unroll
    for (int i = 0; i < 8; ++i) { vals[i] = __expf(vals[i] - mx); sum += vals[i]; }
    red[t] = sum; __syncthreads();
    for (int off = 128; off; off >>= 1) {
        if (t < off) red[t] += red[t + off];
        __syncthreads();
    }
    float inv = 1.0f / red[0];
    #pragma unroll
    for (int i = 0; i < 8; ++i) sc[t + i * 256] = vals[i] * inv;
}

// ---------------------------------------------------------------- kernel 4:
// context partials: grid (dchunk=4, schunk=8, b=32); deterministic (no atomics)
__global__ __launch_bounds__(256) void ctx_partial(const float* __restrict__ E,
                                                   const float* __restrict__ w,
                                                   float* __restrict__ part) {
    const int d  = blockIdx.x * 256 + threadIdx.x;
    const int b  = blockIdx.z, s0 = blockIdx.y * 256;
    const float* wb = w + (size_t)b * S_ + s0;
    const float* Eb = E + ((size_t)b * S_ + s0) * D_ + d;
    float acc = 0.0f;
    #pragma unroll 4
    for (int s = 0; s < 256; ++s) acc += wb[s] * Eb[(size_t)s * D_];
    part[(size_t)blockIdx.y * (B_ * D_) + b * D_ + d] = acc;
}

__global__ __launch_bounds__(256) void ctx_reduce(const float* __restrict__ part,
                                                  float* __restrict__ out) {
    const int i = blockIdx.x * 256 + threadIdx.x;   // 32768
    float acc = 0.0f;
    #pragma unroll
    for (int j = 0; j < 8; ++j) acc += part[(size_t)j * (B_ * D_) + i];
    out[i] = acc;
}

// ---------------------------------------------------------------- launch
extern "C" void kernel_launch(void* const* d_in, const int* in_sizes, int n_in,
                              void* d_out, int out_size, void* d_ws, size_t ws_size,
                              hipStream_t stream) {
    const float* q    = (const float*)d_in[0];   // (32,1024)
    const float* E    = (const float*)d_in[1];   // (32,2048,1024)
    const int*   mask = (const int*)  d_in[2];   // (32,2048)
    const float* Wc   = (const float*)d_in[3];   // (1024,2048)
    const float* Vv   = (const float*)d_in[4];   // (1,1024)

    float* out = (float*)d_out;
    float* ctx = out;                 // 32*1024 floats
    float* wts = out + B_ * D_;       // 32*2048 floats (raw scores -> weights)

    // workspace layout
    unsigned* WeB = (unsigned*)d_ws;                                   // 2 MB
    float* qp   = (float*)((char*)d_ws + (size_t)D_ * D_ * 2);         // 128 KB
    float* part = qp + B_ * D_;                                        // 1 MB

    conv_we<<<(D_ * D_ / 2) / 256, 256, 0, stream>>>(Wc, WeB);
    qprojk<<<D_ / 8, 256, 0, stream>>>(q, Wc, qp);

    const size_t smem = (size_t)64 * LDSK * 2 + 128 * sizeof(float);   // ~132.6 KB
    attnk<<<(B_ * S_) / 64, 256, smem, stream>>>(E, (const unsigned short*)WeB, qp, Vv, wts);

    softmaxk<<<B_, 256, 0, stream>>>(mask, wts);

    ctx_partial<<<dim3(D_ / 256, 8, B_), 256, 0, stream>>>(E, wts, part);
    ctx_reduce<<<(B_ * D_) / 256, 256, 0, stream>>>(part, ctx);
}
